// PrunableLlamaAttention_38001870635106
// MI455X (gfx1250) — matrix operand order repaired and checked by reading, not compiled, not run on hardware
//
#include <hip/hip_runtime.h>
#include <hip/hip_bf16.h>

#define S_LEN 2048
#define HID   4096
#define NH    32
#define NKV   8
#define HD    128
#define GQ    (NH / NKV)

typedef __attribute__((ext_vector_type(16))) __bf16 v16bf;
typedef __attribute__((ext_vector_type(8)))  __bf16 bf16x8;
typedef __attribute__((ext_vector_type(8)))  float  v8f;

union AFrag { v16bf v; bf16x8 h[2]; };

static __device__ __forceinline__ v8f wmma_bf16(v16bf a, v16bf b, v8f c) {
  // D = A(16x32 bf16) * B(32x16 bf16) + C(16x16 f32)
  return __builtin_amdgcn_wmma_f32_16x16x32_bf16(false, a, false, b, (short)0, c, false, false);
}

// ---------------------------------------------------------------- convert
__global__ __launch_bounds__(256) void f32_to_bf16_k(const float* __restrict__ x,
                                                     __bf16* __restrict__ y, int n) {
  int i = (blockIdx.x * blockDim.x + threadIdx.x) * 4;
  if (i + 3 < n) {
    float4 v = *(const float4*)(x + i);
    y[i + 0] = (__bf16)v.x;
    y[i + 1] = (__bf16)v.y;
    y[i + 2] = (__bf16)v.z;
    y[i + 3] = (__bf16)v.w;
  }
}

// ---------------------------------------------------------------- GEMM: Y = A(M x 4096) @ W^T, W is (N x 4096) row-major
// Block: 256 thr = 8 waves. Block tile: 64 rows x 128 cols, K unrolled by 64,
// double-buffered LDS A-stage (8KB x2), ONE barrier per K-step.
// All 4 B-frags + 4 A-frags are loaded BEFORE the 8-WMMA chain so the loads
// land in one clause and overlap the matrix pipe.
// MODE 0: V  -> write transposed  Vt[col][s]            (bf16)
// MODE 1: Q  -> RoPE, write Qr[h][s][d]                 (bf16)
// MODE 2: K  -> RoPE, write Kr[kvh][s][d]               (bf16)
// MODE 3: O  -> write fp32 out[s][col]
template <int MODE>
__global__ __launch_bounds__(256) void gemm_wmma_k(const __bf16* __restrict__ A,
                                                   const __bf16* __restrict__ W,
                                                   void* __restrict__ outp) {
  __shared__ __align__(32) __bf16 As[2][64 * 64];   // 2 x 8KB ping-pong

  const int tid   = threadIdx.x;
  const int wave  = tid >> 5;
  const int lane  = tid & 31;
  const int lhalf = lane >> 4;
  const int l16   = lane & 15;

  const int mBlock = blockIdx.x * 64;
  const int nGroup = blockIdx.y * 128;
  const int mWave  = (wave >> 2) * 32;       // 0 or 32
  const int p      = wave & 3;
  const int loCol  = nGroup + p * 16;
  const int hiCol  = loCol + 64;

  v8f acc[2][2];                              // [m-subtile][lo/hi col tile]
#pragma unroll
  for (int t = 0; t < 2; ++t) { acc[t][0] = (v8f){}; acc[t][1] = (v8f){}; }

  // staging: 256 threads x 32B covers 64 rows x 64 k (8KB)
  const int sRow = tid >> 2;                  // 0..63
  const int sCol = (tid & 3) * 16;            // 0,16,32,48
  const __bf16* aStage = A + (size_t)(mBlock + sRow) * HID + sCol;
  const __bf16* wLo = W + (size_t)(loCol + l16) * HID + lhalf * 16;
  const __bf16* wHi = W + (size_t)(hiCol + l16) * HID + lhalf * 16;

  // prologue: stage first tile
  *(v16bf*)&As[0][sRow * 64 + sCol] = *(const v16bf*)(aStage);

  for (int kk = 0; kk < HID; kk += 64) {
    const int cur = (kk >> 6) & 1;
    __syncthreads();                          // prior stores visible, prior reads done
    if (kk + 64 < HID) {                      // prefetch next tile into other buffer
      *(v16bf*)&As[cur ^ 1][sRow * 64 + sCol] = *(const v16bf*)(aStage + kk + 64);
    }

    // ---- load everything first (one clause of 8x global b128 + 8x ds b128)
    v16bf b[2][2];                            // [c-chunk][lo/hi]
#pragma unroll
    for (int c = 0; c < 2; ++c) {
      b[c][0] = *(const v16bf*)(wLo + kk + c * 32);
      b[c][1] = *(const v16bf*)(wHi + kk + c * 32);
    }
    AFrag a[2][2];                            // [m-subtile][c-chunk]
#pragma unroll
    for (int t = 0; t < 2; ++t) {
      const int mloc = mWave + t * 16 + l16;
#pragma unroll
      for (int c = 0; c < 2; ++c) {
        a[t][c].h[0] = *(const bf16x8*)&As[cur][mloc * 64 + c * 32 + lhalf * 16];
        a[t][c].h[1] = *(const bf16x8*)&As[cur][mloc * 64 + c * 32 + lhalf * 16 + 8];
      }
    }

    // ---- 8 back-to-back WMMAs
#pragma unroll
    for (int c = 0; c < 2; ++c) {
#pragma unroll
      for (int t = 0; t < 2; ++t) {
        acc[t][0] = wmma_bf16(a[t][c].v, b[c][0], acc[t][0]);
        acc[t][1] = wmma_bf16(a[t][c].v, b[c][1], acc[t][1]);
      }
    }
  }

#pragma unroll
  for (int t = 0; t < 2; ++t) {
#pragma unroll
    for (int r = 0; r < 8; ++r) {
      const int s  = mBlock + mWave + t * 16 + r + 8 * lhalf;
      const float lo = acc[t][0][r];
      const float hi = acc[t][1][r];
      if (MODE == 0) {  // V transposed: Vt[col][s]
        __bf16* Vt = (__bf16*)outp;
        Vt[(size_t)(loCol + l16) * S_LEN + s] = (__bf16)lo;
        Vt[(size_t)(hiCol + l16) * S_LEN + s] = (__bf16)hi;
      } else if (MODE == 1 || MODE == 2) {  // RoPE then [head][s][d]
        const int h   = blockIdx.y;           // 128 cols per head
        const int dLo = p * 16 + l16;         // 0..63
        const float invf = __expf(-(float)dLo * 0.14391157f);  // ln(1e4)/64
        const float ang  = (float)s * invf;
        const float c = __cosf(ang), sn = __sinf(ang);
        const float olo = lo * c - hi * sn;   // d<64 : x*cos - x[d+64]*sin
        const float ohi = hi * c + lo * sn;   // d>=64: x*cos + x[d-64]*sin
        __bf16* Q = (__bf16*)outp;
        const size_t base = ((size_t)h * S_LEN + s) * HD;
        Q[base + dLo]      = (__bf16)olo;
        Q[base + dLo + 64] = (__bf16)ohi;
      } else {  // fp32 final output [s][col]
        float* O = (float*)outp;
        O[(size_t)s * HID + loCol + l16] = lo;
        O[(size_t)s * HID + hiCol + l16] = hi;
      }
    }
  }
}

// ---------------------------------------------------------------- flash attention
// One wave per (head, 16-row q tile). Streams keys 32 at a time.
// All 8 K B-frags are preloaded before the score WMMAs; V B-frags preloaded
// in batches of 4 before the P@V WMMAs.
__global__ __launch_bounds__(32) void attn_fa_k(const __bf16* __restrict__ Qr,
                                                const __bf16* __restrict__ Kr,
                                                const __bf16* __restrict__ Vt,
                                                __bf16* __restrict__ Ao) {
  __shared__ __align__(16) __bf16 P[16 * 32];  // one wave per block -> private

  const int lane  = threadIdx.x;
  const int lhalf = lane >> 4;
  const int l16   = lane & 15;
  const int qt    = blockIdx.x;
  const int h     = blockIdx.y;
  const int kvh   = h / GQ;
  const int qbase = qt * 16;
  const float scale = 0.08838834764831845f;  // 1/sqrt(128)

  // Q fragments for the whole 16x128 tile (4 x 16x32 A-frags), kept in VGPRs
  AFrag qf[4];
  const __bf16* qrow = Qr + ((size_t)h * S_LEN + qbase + l16) * HD + lhalf * 16;
#pragma unroll
  for (int f = 0; f < 4; ++f) {
    qf[f].h[0] = *(const bf16x8*)(qrow + f * 32);
    qf[f].h[1] = *(const bf16x8*)(qrow + f * 32 + 8);
  }

  float mi[8], li[8];
  v8f Oacc[8];
#pragma unroll
  for (int r = 0; r < 8; ++r) { mi[r] = -3.0e38f; li[r] = 0.0f; }
#pragma unroll
  for (int t = 0; t < 8; ++t) Oacc[t] = (v8f){};

  const int kEnd = qbase + 16;
  for (int kb = 0; kb < kEnd; kb += 32) {
    const __bf16* kp0 = Kr + ((size_t)kvh * S_LEN + kb + l16) * HD + lhalf * 16;
    const __bf16* kp1 = kp0 + (size_t)16 * HD;
    const __bf16* vp  = Vt + ((size_t)kvh * HD + l16) * S_LEN + kb + lhalf * 16;

    if (kb + 32 < kEnd) {                     // keep the K/V stream ahead
      __builtin_prefetch(kp0 + (size_t)32 * HD, 0, 1);
      __builtin_prefetch(kp1 + (size_t)32 * HD, 0, 1);
      __builtin_prefetch(vp + 32, 0, 1);
    }

    // ---- preload all 8 K B-frags (16x global_load_b128 in one clause)
    v16bf kbf[8];
#pragma unroll
    for (int f = 0; f < 4; ++f) {
      kbf[f]     = *(const v16bf*)(kp0 + f * 32);
      kbf[4 + f] = *(const v16bf*)(kp1 + f * 32);
    }

    // scores: S[16q x 32k] as two 16x16 C tiles
    v8f sc0 = {}, sc1 = {};
#pragma unroll
    for (int f = 0; f < 4; ++f) {
      sc0 = wmma_bf16(qf[f].v, kbf[f], sc0);
      sc1 = wmma_bf16(qf[f].v, kbf[4 + f], sc1);
    }

    // online softmax; row m = r + 8*lhalf lives in one half-wave (cols = l16)
#pragma unroll
    for (int r = 0; r < 8; ++r) {
      const int qg  = qbase + r + 8 * lhalf;
      const int kg0 = kb + l16;
      const int kg1 = kb + 16 + l16;
      float s0 = (kg0 <= qg) ? sc0[r] * scale : -1.0e30f;
      float s1 = (kg1 <= qg) ? sc1[r] * scale : -1.0e30f;
      float mx = fmaxf(s0, s1);
#pragma unroll
      for (int off = 1; off < 16; off <<= 1) mx = fmaxf(mx, __shfl_xor(mx, off, 32));
      const float mnew  = fmaxf(mi[r], mx);
      const float alpha = __expf(mi[r] - mnew);
      const float p0 = __expf(s0 - mnew);
      const float p1 = __expf(s1 - mnew);
      float ps = p0 + p1;
#pragma unroll
      for (int off = 1; off < 16; off <<= 1) ps += __shfl_xor(ps, off, 32);
      li[r] = li[r] * alpha + ps;
      mi[r] = mnew;
#pragma unroll
      for (int t = 0; t < 8; ++t) Oacc[t][r] *= alpha;
      const int m = r + 8 * lhalf;
      P[m * 32 + l16]      = (__bf16)p0;   // C-layout -> row-major P in LDS
      P[m * 32 + 16 + l16] = (__bf16)p1;
    }
    asm volatile("" ::: "memory");  // LDS ops from one wave retire in order

    // reload P as A-frag (16x32), then O += P @ V  (8 dim tiles, batched 4+4)
    AFrag pf;
    pf.h[0] = *(const bf16x8*)&P[l16 * 32 + lhalf * 16];
    pf.h[1] = *(const bf16x8*)&P[l16 * 32 + lhalf * 16 + 8];
#pragma unroll
    for (int half = 0; half < 2; ++half) {
      v16bf vb[4];
#pragma unroll
      for (int t = 0; t < 4; ++t) {
        vb[t] = *(const v16bf*)(vp + (size_t)(half * 4 + t) * 16 * S_LEN);
      }
#pragma unroll
      for (int t = 0; t < 4; ++t) {
        Oacc[half * 4 + t] = wmma_bf16(pf.v, vb[t], Oacc[half * 4 + t]);
      }
    }
  }

  // normalize and store attn output row-major [s][h*128+d] for the O-proj GEMM
#pragma unroll
  for (int r = 0; r < 8; ++r) {
    const int s = qbase + r + 8 * lhalf;
    const float rinv = 1.0f / li[r];
#pragma unroll
    for (int t = 0; t < 8; ++t) {
      Ao[(size_t)s * (NH * HD) + (size_t)h * HD + t * 16 + l16] = (__bf16)(Oacc[t][r] * rinv);
    }
  }
}

// ---------------------------------------------------------------- launcher
extern "C" void kernel_launch(void* const* d_in, const int* in_sizes, int n_in,
                              void* d_out, int out_size, void* d_ws, size_t ws_size,
                              hipStream_t stream) {
  (void)in_sizes; (void)n_in; (void)out_size; (void)ws_size;
  const float* hidden = (const float*)d_in[0];
  // d_in[1] = attention_mask (causal, implemented analytically), d_in[2] = position_ids (= arange)
  const float* Wq = (const float*)d_in[3];
  const float* Wk = (const float*)d_in[4];
  const float* Wv = (const float*)d_in[5];
  const float* Wo = (const float*)d_in[6];

  char* ws = (char*)d_ws;
  __bf16* Xb  = (__bf16*)(ws);                           // 16 MB  S x HID
  __bf16* Wqb = (__bf16*)(ws + (size_t)16  * (1 << 20)); // 32 MB
  __bf16* Wkb = (__bf16*)(ws + (size_t)48  * (1 << 20)); //  8 MB
  __bf16* Wvb = (__bf16*)(ws + (size_t)56  * (1 << 20)); //  8 MB
  __bf16* Wob = (__bf16*)(ws + (size_t)64  * (1 << 20)); // 32 MB
  __bf16* Qr  = (__bf16*)(ws + (size_t)96  * (1 << 20)); // 16 MB  [NH][S][HD] (RoPE'd)
  __bf16* Kr  = (__bf16*)(ws + (size_t)112 * (1 << 20)); //  4 MB  [NKV][S][HD] (RoPE'd)
  __bf16* Vt  = (__bf16*)(ws + (size_t)116 * (1 << 20)); //  4 MB  [NKV*HD][S]
  __bf16* Ao  = (__bf16*)(ws + (size_t)120 * (1 << 20)); // 16 MB  [S][NH*HD]

  const int nX = S_LEN * HID;
  const int nQ = NH * HD * HID;
  const int nK = NKV * HD * HID;
  f32_to_bf16_k<<<nX / 1024, 256, 0, stream>>>(hidden, Xb, nX);
  f32_to_bf16_k<<<nQ / 1024, 256, 0, stream>>>(Wq, Wqb, nQ);
  f32_to_bf16_k<<<nK / 1024, 256, 0, stream>>>(Wk, Wkb, nK);
  f32_to_bf16_k<<<nK / 1024, 256, 0, stream>>>(Wv, Wvb, nK);
  f32_to_bf16_k<<<nQ / 1024, 256, 0, stream>>>(Wo, Wob, nQ);

  gemm_wmma_k<1><<<dim3(S_LEN / 64, (NH  * HD) / 128), 256, 0, stream>>>(Xb, Wqb, Qr);
  gemm_wmma_k<2><<<dim3(S_LEN / 64, (NKV * HD) / 128), 256, 0, stream>>>(Xb, Wkb, Kr);
  gemm_wmma_k<0><<<dim3(S_LEN / 64, (NKV * HD) / 128), 256, 0, stream>>>(Xb, Wvb, Vt);

  attn_fa_k<<<dim3(S_LEN / 16, NH), 32, 0, stream>>>(Qr, Kr, Vt, Ao);

  gemm_wmma_k<3><<<dim3(S_LEN / 64, HID / 128), 256, 0, stream>>>(Ao, Wob, d_out);
}
